// LSTM_Turnover_53850299957977
// MI455X (gfx1250) — matrix operand order repaired and checked by reading, not compiled, not run on hardware
//
#include <hip/hip_runtime.h>
#include <hip/hip_bf16.h>

typedef _Float16 v16h __attribute__((ext_vector_type(16)));
typedef float    v8f  __attribute__((ext_vector_type(8)));

#define HDIM 64
#define GH   256   // 4*HDIM
#define WARM 15

union AF { uint4 u[2]; v16h h; };

// WMMA: D = A(16x32 f16) * B(32x16 f16) + C(16x16 f32)
__device__ __forceinline__ v8f wmma_f16(v16h a, v16h b, v8f c) {
  return __builtin_amdgcn_wmma_f32_16x16x32_f16(false, a, false, b, (short)0, c, false, false);
}

// A fragment from LDS tile stored row-major [16][64] f16.
// Layout (ISA 7.12.2, 16-bit A 16x32): lanes 0-15 row=lane get K blocks [0..7],[16..23];
// lanes 16-31 row=lane-16 get K blocks [8..15],[24..31].
__device__ __forceinline__ v16h load_a(const _Float16* tile, int kt, int r, int hi) {
  const _Float16* p = tile + r * HDIM + kt * 32;
  AF f;
  f.u[0] = *(const uint4*)(p + hi * 8);
  f.u[1] = *(const uint4*)(p + 16 + hi * 8);
  return f.h;
}

// B fragment from global weight W stored [256][64] f16 row-major (gates = x @ W^T,
// so B(k,n) = W[n][k]; per-lane column n is a contiguous row of W).
// Layout: lanes 0-15 col n=lane, K blocks [0..7],[16..23]; lanes 16-31 col n=lane-16, K blocks [8..15],[24..31].
__device__ __forceinline__ v16h load_b(const _Float16* __restrict__ W, int ct, int kt, int nl, int hi) {
  const _Float16* p = W + (ct * 16 + nl) * HDIM + kt * 32;
  AF f;
  f.u[0] = *(const uint4*)(p + hi * 8);
  f.u[1] = *(const uint4*)(p + 16 + hi * 8);
  return f.h;
}

// Branch-free fast activations (v_exp_f32 + v_rcp_f32; exact saturation at +/-inf).
__device__ __forceinline__ float sigm(float v) {
  return __builtin_amdgcn_rcpf(1.f + __expf(-v));
}
__device__ __forceinline__ float tanh_fast(float v) {
  // tanh(x) = 1 - 2/(exp(2x)+1); exp->inf => 1, exp->0 => -1.
  return __builtin_fmaf(-2.f, __builtin_amdgcn_rcpf(__expf(2.f * v) + 1.f), 1.f);
}

// ---------------- prep: convert weights to f16, fold biases ----------------
__global__ void lstm_prep(const float* __restrict__ Wih0, const float* __restrict__ Whh0,
                          const float* __restrict__ bih0, const float* __restrict__ bhh0,
                          const float* __restrict__ Wih,  const float* __restrict__ Whh,
                          const float* __restrict__ bih,  const float* __restrict__ bhh,
                          const float* __restrict__ Wlin, const float* __restrict__ blin_in,
                          _Float16* __restrict__ whh_h, _Float16* __restrict__ wih_h,
                          float* __restrict__ bsum, float* __restrict__ wih0f,
                          float* __restrict__ wlinf, float* __restrict__ blinf) {
  const int NW = 4 * GH * HDIM;   // 65536: whh layers 0..3
  const int NI = 3 * GH * HDIM;   // 49152: wih layers 1..3
  int i = blockIdx.x * blockDim.x + threadIdx.x;
  if (i < NW) {
    int l = i >> 14, j = i & 16383;
    float v = (l == 0) ? Whh0[j] : Whh[(l - 1) * 16384 + j];
    whh_h[i] = (_Float16)v;
  } else if (i < NW + NI) {
    int k = i - NW;
    wih_h[16384 + k] = (_Float16)Wih[k];          // slot 0 unused
  } else if (i < NW + NI + 1024) {
    int k = i - NW - NI;
    int l = k >> 8, j = k & 255;
    bsum[k] = (l == 0) ? (bih0[j] + bhh0[j]) : (bih[(l - 1) * 256 + j] + bhh[(l - 1) * 256 + j]);
  } else if (i < NW + NI + 1024 + 512) {
    int k = i - (NW + NI + 1024);
    wih0f[k] = Wih0[k];
  } else if (i < NW + NI + 1024 + 512 + 64) {
    int k = i - (NW + NI + 1024 + 512);
    wlinf[k] = Wlin[k];
  } else if (i == NW + NI + 1024 + 512 + 64) {
    blinf[0] = blin_in[0];
  }
}

// ---------------- main: each wave owns a 16-row tile for all 23 steps ----------------
__global__ __launch_bounds__(64) void lstm_main(
    const float* __restrict__ x, const float* __restrict__ ax, const float* __restrict__ ay,
    const _Float16* __restrict__ wih_h, const _Float16* __restrict__ whh_h,
    const float* __restrict__ bsum, const float* __restrict__ wih0,
    const float* __restrict__ wlin, const float* __restrict__ blin,
    const int* __restrict__ fp_ptr, float* __restrict__ out) {
  __shared__ _Float16 hbuf[2][4][16][HDIM];  // per-wave h state, f16 row-major (A operand)
  __shared__ float    cbuf[2][4][16][HDIM];  // per-wave c state, f32
  __shared__ float    x01[2][16][2];         // per-row layer-0 input (x0, x1)

  const int tid  = threadIdx.x;
  const int wv   = tid >> 5;
  const int lane = tid & 31;
  const int nl   = lane & 15;
  const int hi   = lane >> 4;                // 0: rows 0..7 of C-frag, 1: rows 8..15
  const int rowbase = blockIdx.x * 32 + wv * 16;

  // zero this wave's state (wave-local, no barrier needed)
  for (int i = lane; i < 4 * 16 * HDIM; i += 32) {
    ((_Float16*)hbuf[wv])[i] = (_Float16)0.f;
    ((float*)cbuf[wv])[i] = 0.f;
  }
  asm volatile("s_wait_dscnt 0x0" ::: "memory");

  int fpc = fp_ptr[0];
  fpc = fpc < 0 ? 0 : (fpc > 8 ? 8 : fpc);
  const int steps = WARM + fpc;
  const int ow = steps;
  const float bl = blin[0];
  float prev_out = 0.f;

  for (int t = 0; t < steps; ++t) {
    // ---- layer-0 inputs for this step (lanes 0..15 write their row) ----
    if (hi == 0) {
      int rg = rowbase + nl;
      float x0, x1;
      if (t < WARM) { x0 = x[rg * 16 + t];  x1 = ax[rg * 16 + t + 1]; }
      else          { x0 = (t == WARM) ? x[rg * 16 + 15] : prev_out;
                      x1 = ay[rg * 8 + (t - WARM)]; }
      x01[wv][nl][0] = x0;
      x01[wv][nl][1] = x1;
    }
    asm volatile("s_wait_dscnt 0x0" ::: "memory");

    // ---- layer 0: gates = [x0,x1] @ Wih0^T + h0 @ Whh0^T + b ----
    {
      const _Float16* ht = &hbuf[wv][0][0][0];
      v16h ah0 = load_a(ht, 0, nl, hi);
      v16h ah1 = load_a(ht, 1, nl, hi);
      float xm0[8], xm1[8];
#pragma unroll
      for (int vv = 0; vv < 8; ++vv) {
        int m = vv + hi * 8;
        xm0[vv] = x01[wv][m][0];
        xm1[vv] = x01[wv][m][1];
      }
#pragma unroll
      for (int jt = 0; jt < 4; ++jt) {
        v8f acc[4];
#pragma unroll
        for (int g = 0; g < 4; ++g) {
          int ct = g * 4 + jt;
          int n0 = ct * 16 + nl;
          float b  = bsum[n0];
          float w0 = wih0[n0 * 2 + 0];
          float w1 = wih0[n0 * 2 + 1];
          v8f a;
#pragma unroll
          for (int vv = 0; vv < 8; ++vv) a[vv] = b + xm0[vv] * w0 + xm1[vv] * w1;
          a = wmma_f16(ah0, load_b(whh_h, ct, 0, nl, hi), a);
          a = wmma_f16(ah1, load_b(whh_h, ct, 1, nl, hi), a);
          acc[g] = a;
        }
        int j = jt * 16 + nl;
#pragma unroll
        for (int vv = 0; vv < 8; ++vv) {
          int m = vv + hi * 8;
          float ig = sigm(acc[0][vv]);
          float fg = sigm(acc[1][vv]);
          float gg = tanh_fast(acc[2][vv]);
          float og = sigm(acc[3][vv]);
          float cn = fg * cbuf[wv][0][m][j] + ig * gg;
          cbuf[wv][0][m][j] = cn;
          hbuf[wv][0][m][j] = (_Float16)(og * tanh_fast(cn));
        }
      }
    }

    // ---- layers 1..3: gates = h_{l-1} @ Wih^T + h_l @ Whh^T + b ----
    for (int l = 1; l < 4; ++l) {
      const _Float16* xt = &hbuf[wv][l - 1][0][0];
      const _Float16* ht = &hbuf[wv][l][0][0];
      v16h ax0 = load_a(xt, 0, nl, hi);
      v16h ax1 = load_a(xt, 1, nl, hi);
      v16h ah0 = load_a(ht, 0, nl, hi);
      v16h ah1 = load_a(ht, 1, nl, hi);
      const _Float16* Wi = wih_h + l * GH * HDIM;
      const _Float16* Wh = whh_h + l * GH * HDIM;
      const float* bs = bsum + l * GH;
#pragma unroll
      for (int jt = 0; jt < 4; ++jt) {
        v8f acc[4];
#pragma unroll
        for (int g = 0; g < 4; ++g) {
          int ct = g * 4 + jt;
          float b = bs[ct * 16 + nl];
          v8f a = { b, b, b, b, b, b, b, b };
          a = wmma_f16(ax0, load_b(Wi, ct, 0, nl, hi), a);
          a = wmma_f16(ax1, load_b(Wi, ct, 1, nl, hi), a);
          a = wmma_f16(ah0, load_b(Wh, ct, 0, nl, hi), a);
          a = wmma_f16(ah1, load_b(Wh, ct, 1, nl, hi), a);
          acc[g] = a;
        }
        int j = jt * 16 + nl;
#pragma unroll
        for (int vv = 0; vv < 8; ++vv) {
          int m = vv + hi * 8;
          float ig = sigm(acc[0][vv]);
          float fg = sigm(acc[1][vv]);
          float gg = tanh_fast(acc[2][vv]);
          float og = sigm(acc[3][vv]);
          float cn = fg * cbuf[wv][l][m][j] + ig * gg;
          cbuf[wv][l][m][j] = cn;
          hbuf[wv][l][m][j] = (_Float16)(og * tanh_fast(cn));
        }
      }
    }

    // ---- head: out = x1 + tanh(h3 . Wlin + b) ----
    asm volatile("s_wait_dscnt 0x0" ::: "memory");
    if (hi == 0) {
      int rg = rowbase + nl;
      float acc = bl;
#pragma unroll 8
      for (int j = 0; j < HDIM; ++j)
        acc += (float)hbuf[wv][3][nl][j] * wlin[j];
      float o = x01[wv][nl][1] + tanh_fast(acc);
      out[rg * ow + t] = o;
      prev_out = o;
    }
  }
}

extern "C" void kernel_launch(void* const* d_in, const int* in_sizes, int n_in,
                              void* d_out, int out_size, void* d_ws, size_t ws_size,
                              hipStream_t stream) {
  const float* x    = (const float*)d_in[0];
  const float* ax   = (const float*)d_in[1];
  const float* ay   = (const float*)d_in[2];
  const float* Wih0 = (const float*)d_in[3];
  const float* Whh0 = (const float*)d_in[4];
  const float* bih0 = (const float*)d_in[5];
  const float* bhh0 = (const float*)d_in[6];
  const float* Wih  = (const float*)d_in[7];
  const float* Whh  = (const float*)d_in[8];
  const float* bih  = (const float*)d_in[9];
  const float* bhh  = (const float*)d_in[10];
  const float* Wlin = (const float*)d_in[11];
  const float* blin = (const float*)d_in[12];
  const int*   fp   = (const int*)d_in[13];

  char* ws = (char*)d_ws;
  _Float16* whh_h = (_Float16*)(ws);             // 4*256*64 f16 = 131072 B
  _Float16* wih_h = (_Float16*)(ws + 131072);    // 4*256*64 f16 = 131072 B (slot 0 unused)
  float* bsum  = (float*)(ws + 262144);          // 4*256 f32
  float* wih0f = (float*)(ws + 266240);          // 256*2 f32
  float* wlinf = (float*)(ws + 268288);          // 64 f32
  float* blinf = (float*)(ws + 268544);          // 1 f32

  const int Ntot = 65536 + 49152 + 1024 + 512 + 64 + 1;
  lstm_prep<<<(Ntot + 255) / 256, 256, 0, stream>>>(
      Wih0, Whh0, bih0, bhh0, Wih, Whh, bih, bhh, Wlin, blin,
      whh_h, wih_h, bsum, wih0f, wlinf, blinf);

  int B = in_sizes[0] / 16;                      // 32768
  int blocks = B / 32;                           // 32 rows per block (2 waves x 16)
  lstm_main<<<blocks, 64, 0, stream>>>(
      x, ax, ay, wih_h, whh_h, bsum, wih0f, wlinf, blinf, fp, (float*)d_out);
}
